// AttentionGuidedMemory_29566554866090
// MI455X (gfx1250) — matrix operand order repaired and checked by reading, not compiled, not run on hardware
//
#include <hip/hip_runtime.h>
#include <hip/hip_bf16.h>
#include <cstdint>
#include <cstddef>

typedef __bf16 bf16;
typedef __attribute__((ext_vector_type(16))) __bf16 v16bf;
typedef __attribute__((ext_vector_type(8)))  __bf16 v8bf;
typedef __attribute__((ext_vector_type(8)))  float  v8f;

// problem dims
constexpr int BB = 512;   // batch
constexpr int SS = 1024;  // seq len
constexpr int KD = 50;    // key dim
constexpr int MD = 50;    // memory slots
constexpr int VD = 200;   // value dim
constexpr int AD = 64;    // attention feature dim
// padded dims
constexpr int KP  = 64;   // K padded to 2x32
constexpr int MP  = 64;   // M padded
constexpr int VP  = 224;  // V padded to 7x32 (WMMA K for rc GEMM)
constexpr int CW  = 128;  // concat width 2A (exact 4x32)
constexpr int NPV = 208;  // V padded to 13x16 (WMMA N for erase/add)
constexpr int NB  = 16;   // batch rows per block

// swizzled weight segments (bf16 element offsets in d_ws)
constexpr int SWZ_WC = 0;      // 2 ktiles x 4 ntiles
constexpr int SWZ_KM = 4096;   // 2 x 4
constexpr int SWZ_WR = 8192;   // 7 x 4
constexpr int SWZ_WE = 22528;  // 4 x 13
constexpr int SWZ_WA = 49152;  // 4 x 13
constexpr int SWZ_TOTAL = 75776;

// ---------------------------------------------------------------------------
// Fast transcendental helpers -> single CDNA5 trans ops (v_exp_f32 = 2^x,
// v_tanh_f32, v_rcp_f32), no OCML branchy range fix-ups.
// ---------------------------------------------------------------------------
constexpr float LOG2E = 1.4426950408889634f;

__device__ __forceinline__ float fexp2(float x) {
#if __has_builtin(__builtin_amdgcn_exp2f)
  return __builtin_amdgcn_exp2f(x);
#else
  return exp2f(x);
#endif
}
__device__ __forceinline__ float frcp(float x) {
#if __has_builtin(__builtin_amdgcn_rcpf)
  return __builtin_amdgcn_rcpf(x);
#else
  return 1.0f / x;
#endif
}
__device__ __forceinline__ float ftanh(float x) {
#if __has_builtin(__builtin_amdgcn_tanhf)
  return __builtin_amdgcn_tanhf(x);
#else
  float e = fexp2(2.0f * LOG2E * x);          // e^(2x)
  return 1.0f - 2.0f * frcp(e + 1.0f);
#endif
}
__device__ __forceinline__ float fsigmoid(float x) {
  return frcp(1.0f + fexp2(-LOG2E * x));
}

// ---------------------------------------------------------------------------
// Prep: swizzle W^T (B-matrix of each GEMM) into the per-lane WMMA bf16 layout
// B fragment layout (16-bit 32x16): lane l<16 -> column N=l, K=e (e<8), K=8+e (e>=8);
// lanes 16..31 -> column N=l-16, K=8+e (e<8), K=16+e (e>=8).  Tile = kk*NT+nt, 32 lanes x 16 bf16.
// ---------------------------------------------------------------------------
__global__ void agm_prep_swizzle(const float* __restrict__ Wc,
                                 const float* __restrict__ Km,
                                 const float* __restrict__ Wr,
                                 const float* __restrict__ We,
                                 const float* __restrict__ Wa,
                                 unsigned short* __restrict__ swz_u) {
  int idx = blockIdx.x * 256 + threadIdx.x;
  if (idx >= SWZ_TOTAL) return;
  bf16* swz = reinterpret_cast<bf16*>(swz_u);

  const float* W; int Nw, Kw, NT, off;
  if (idx < SWZ_KM)      { W = Wc; Nw = KD;  Kw = KD;  NT = 4;  off = SWZ_WC; }
  else if (idx < SWZ_WR) { W = Km; Nw = MD;  Kw = KD;  NT = 4;  off = SWZ_KM; }
  else if (idx < SWZ_WE) { W = Wr; Nw = AD;  Kw = VD;  NT = 4;  off = SWZ_WR; }
  else if (idx < SWZ_WA) { W = We; Nw = VD;  Kw = CW;  NT = 13; off = SWZ_WE; }
  else                   { W = Wa; Nw = VD;  Kw = CW;  NT = 13; off = SWZ_WA; }

  int li   = idx - off;
  int e    = li & 15;          // element within lane fragment
  int lane = (li >> 4) & 31;   // lane
  int tile = li >> 9;          // kk*NT + nt
  int kk   = tile / NT;
  int nt   = tile % NT;
  int half = lane >> 4;
  int n    = nt * 16 + (lane & 15);
  int k    = kk * 32 + (e >> 3) * 16 + half * 8 + (e & 7);  // B-matrix K index for this element
  float v  = (n < Nw && k < Kw) ? W[(size_t)n * Kw + k] : 0.0f;  // B[k][n]=W[n][k]
  swz[idx] = (bf16)v;
}

// ---------------------------------------------------------------------------
// One 16xN-tile GEMM column-strip: acc(16x16 f32) += A(16 x 32*KT) * B-tile.
// A is bf16 in LDS, row-major with stride lda (multiple of 16B); B pre-swizzled.
// A fragment layout (16-bit 16x32): lane l -> row M=l&15, half=l>>4;
// elements 0..7 -> K = k0+half*8+e ; elements 8..15 -> K = k0+16+half*8+(e-8).
// ---------------------------------------------------------------------------
__device__ __forceinline__ v8f wmma_strip(const bf16* __restrict__ A, int lda,
                                          const bf16* __restrict__ Bswz,
                                          int NT, int nt, int KT, int lane) {
  const int half = lane >> 4;
  const int r    = lane & 15;
  v8f acc = {0.f, 0.f, 0.f, 0.f, 0.f, 0.f, 0.f, 0.f};
  for (int kk = 0; kk < KT; ++kk) {
    const bf16* a0 = A + r * lda + kk * 32 + half * 8;
    v8bf A0 = *reinterpret_cast<const v8bf*>(a0);        // K = base..base+7
    v8bf A1 = *reinterpret_cast<const v8bf*>(a0 + 16);   // K = base+16..base+23
    v16bf av = __builtin_shufflevector(A0, A1, 0,1,2,3,4,5,6,7,
                                               8,9,10,11,12,13,14,15);
    const bf16* bp = Bswz + (((size_t)(kk * NT + nt)) * 32 + lane) * 16;
    v8bf B0 = *reinterpret_cast<const v8bf*>(bp);
    v8bf B1 = *reinterpret_cast<const v8bf*>(bp + 8);
    v16bf bv = __builtin_shufflevector(B0, B1, 0,1,2,3,4,5,6,7,
                                               8,9,10,11,12,13,14,15);
    acc = __builtin_amdgcn_wmma_f32_16x16x32_bf16(false, av, false, bv,
                                                  (short)0, acc, false, false);
  }
  return acc;
}

// ---------------------------------------------------------------------------
// Persistent recurrent kernel: 1 block = 16 batch rows, loops over all t.
// Memory state streamed as float4 (global b128, fully coalesced, L2-resident).
// ---------------------------------------------------------------------------
__global__ __launch_bounds__(256)
void agm_scan_kernel(const float* __restrict__ qptr,   // (B,S,K)
                     const float* __restrict__ aptr,   // (B,S,A)
                     const float* __restrict__ bc,
                     const float* __restrict__ br,
                     const float* __restrict__ be,
                     const float* __restrict__ ba,
                     const unsigned short* __restrict__ swz_u,
                     float* __restrict__ rc_out,       // (B,S,A)
                     float* __restrict__ mem,          // (B,M,V) in-place state
                     float* __restrict__ w_out) {      // (B,S,M)
  const bf16* swz = reinterpret_cast<const bf16*>(swz_u);

  __shared__ alignas(16) bf16  s_q [NB * KP];    // bf16 q_t, K-padded
  __shared__ alignas(16) bf16  s_ck[NB * KP];    // tanh(q Wc^T + bc)
  __shared__ alignas(16) bf16  s_rw[NB * VP];    // weighted memory read, V-padded
  __shared__ alignas(16) bf16  s_wi[NB * CW];    // [a_t , rc]
  __shared__ alignas(16) float s_w [NB * MP];    // scores -> softmax weights
  __shared__ alignas(16) float s_er[NB * NPV];   // erase gate
  __shared__ alignas(16) float s_ad[NB * NPV];   // add vector
  __shared__ alignas(16) float s_bias[64 + 64 + 208 + 208];  // bc|br|be|ba

  const int tid  = threadIdx.x;
  const int wave = tid >> 5;
  const int lane = tid & 31;
  const int n16  = lane & 15;
  const int mb   = (lane >> 4) << 3;      // D-fragment row base (0 or 8)
  const int b0   = blockIdx.x * NB;

  // one-time: biases (zero-padded) and static zero pads of bf16 staging
  for (int i = tid; i < 64 + 64 + 208 + 208; i += 256) {
    float v = 0.f;
    if (i < 64)        { if (i < KD) v = bc[i]; }
    else if (i < 128)  { v = br[i - 64]; }
    else if (i < 336)  { int c = i - 128; if (c < VD) v = be[c]; }
    else               { int c = i - 336; if (c < VD) v = ba[c]; }
    s_bias[i] = v;
  }
  for (int i = tid; i < NB * (KP - KD); i += 256) {          // s_q pad cols
    int r = i / (KP - KD), c = KD + i % (KP - KD);
    s_q[r * KP + c] = (bf16)0.f;
  }
  for (int i = tid; i < NB * (VP - VD); i += 256) {          // s_rw pad cols
    int r = i / (VP - VD), c = VD + i % (VP - VD);
    s_rw[r * VP + c] = (bf16)0.f;
  }
  __syncthreads();

  // einsum / update thread map: 16 threads per batch row, float4 over V.
  // Sweeps j=0..2 cover v4 = vl+16j (0..47) -> v 0..191; j=3 only vl<2
  // (v4 = 48,49 -> v 192..199).  All accesses 16B-aligned & coalesced.
  const int r16 = tid >> 4;
  const int vl  = tid & 15;
  const bool j3 = (vl < 2);

  for (int t = 0; t < SS; ++t) {
    // ---- stage q_t (bf16) and a_t (bf16, first half of wi) ----
    for (int i = tid; i < NB * KD; i += 256) {
      int r = i / KD, k = i % KD;
      s_q[r * KP + k] = (bf16)qptr[((size_t)(b0 + r) * SS + t) * KD + k];
    }
    for (int i = tid; i < NB * AD; i += 256) {
      int r = i >> 6, c = i & 63;
      s_wi[r * CW + c] = (bf16)aptr[((size_t)(b0 + r) * SS + t) * AD + c];
    }
    if (t + 1 < SS && tid < NB) {   // hint next step's rows toward WGP cache
      __builtin_prefetch(qptr + ((size_t)(b0 + tid) * SS + (t + 1)) * KD, 0, 3);
      __builtin_prefetch(aptr + ((size_t)(b0 + tid) * SS + (t + 1)) * AD, 0, 3);
    }
    __syncthreads();

    // ---- GEMM1: ck = tanh(q @ Wc^T + bc)  (waves 0-3, one 16-col tile each)
    if (wave < 4) {
      v8f c = wmma_strip(s_q, KP, swz + SWZ_WC, 4, wave, 2, lane);
      int col = wave * 16 + n16;
      float bias = s_bias[col];
      #pragma unroll
      for (int j = 0; j < 8; ++j)
        s_ck[(mb + j) * KP + col] = (bf16)ftanh(c[j] + bias);
    }
    __syncthreads();

    // ---- GEMM2: scores = ck @ KM^T ----
    if (wave < 4) {
      v8f c = wmma_strip(s_ck, KP, swz + SWZ_KM, 4, wave, 2, lane);
      int col = wave * 16 + n16;
      #pragma unroll
      for (int j = 0; j < 8; ++j) s_w[(mb + j) * MP + col] = c[j];
    }
    __syncthreads();

    // ---- softmax over m<50, 2 rows per wave, wave32 shuffle reductions ----
    #pragma unroll
    for (int rr = 0; rr < 2; ++rr) {
      int r = wave * 2 + rr;
      float x0 = (lane      < MD) ? s_w[r * MP + lane]      : -3.0e38f;
      float x1 = (lane + 32 < MD) ? s_w[r * MP + lane + 32] : -3.0e38f;
      float mx = fmaxf(x0, x1);
      #pragma unroll
      for (int o = 16; o > 0; o >>= 1) mx = fmaxf(mx, __shfl_xor(mx, o, 32));
      float e0 = (lane      < MD) ? fexp2(LOG2E * (x0 - mx)) : 0.f;
      float e1 = (lane + 32 < MD) ? fexp2(LOG2E * (x1 - mx)) : 0.f;
      float sm = e0 + e1;
      #pragma unroll
      for (int o = 16; o > 0; o >>= 1) sm += __shfl_xor(sm, o, 32);
      float inv = frcp(sm);
      size_t wo = ((size_t)(b0 + r) * SS + t) * MD;
      if (lane < MD)      { float wv = e0 * inv; s_w[r * MP + lane]      = wv; w_out[wo + lane]      = wv; }
      if (lane + 32 < MD) { float wv = e1 * inv; s_w[r * MP + lane + 32] = wv; w_out[wo + lane + 32] = wv; }
    }
    __syncthreads();

    // ---- einsum: rw[r][v] = sum_m w[r][m] * mem[b0+r][m][v]  (b128 stream) ----
    {
      const float4* mrow4 =
          (const float4*)(mem + (size_t)(b0 + r16) * (MD * VD));  // stride 50 f4
      float4 acc[4];
      #pragma unroll
      for (int j = 0; j < 4; ++j) acc[j] = make_float4(0.f, 0.f, 0.f, 0.f);
      for (int m = 0; m < MD; ++m) {
        float wm = s_w[r16 * MP + m];
        const float4* mp = mrow4 + m * (VD / 4);
        #pragma unroll
        for (int j = 0; j < 3; ++j) {
          float4 d = mp[vl + j * 16];
          acc[j].x += wm * d.x; acc[j].y += wm * d.y;
          acc[j].z += wm * d.z; acc[j].w += wm * d.w;
        }
        if (j3) {
          float4 d = mp[vl + 48];
          acc[3].x += wm * d.x; acc[3].y += wm * d.y;
          acc[3].z += wm * d.z; acc[3].w += wm * d.w;
        }
      }
      #pragma unroll
      for (int j = 0; j < 3; ++j) {
        int v = (vl + j * 16) * 4;
        s_rw[r16 * VP + v + 0] = (bf16)acc[j].x;
        s_rw[r16 * VP + v + 1] = (bf16)acc[j].y;
        s_rw[r16 * VP + v + 2] = (bf16)acc[j].z;
        s_rw[r16 * VP + v + 3] = (bf16)acc[j].w;
      }
      if (j3) {
        int v = (vl + 48) * 4;
        s_rw[r16 * VP + v + 0] = (bf16)acc[3].x;
        s_rw[r16 * VP + v + 1] = (bf16)acc[3].y;
        s_rw[r16 * VP + v + 2] = (bf16)acc[3].z;
        s_rw[r16 * VP + v + 3] = (bf16)acc[3].w;
      }
    }
    __syncthreads();

    // ---- GEMM3: rc = rw @ Wr^T + br ; emit output + bf16 into wi[64:128] ----
    if (wave < 4) {
      v8f c = wmma_strip(s_rw, VP, swz + SWZ_WR, 4, wave, 7, lane);
      int col = wave * 16 + n16;
      float bias = s_bias[64 + col];
      #pragma unroll
      for (int j = 0; j < 8; ++j) {
        int r2 = mb + j;
        float v = c[j] + bias;
        rc_out[((size_t)(b0 + r2) * SS + t) * AD + col] = v;
        s_wi[r2 * CW + AD + col] = (bf16)v;
      }
    }
    __syncthreads();

    if (t < SS - 1) {
      // ---- GEMM4/5: erase = sigmoid(wi We^T + be), add = tanh(wi Wa^T + ba)
      for (int job = wave; job < 26; job += 8) {
        bool isE = job < 13;
        int  nt  = isE ? job : job - 13;
        const bf16* wseg = swz + (isE ? SWZ_WE : SWZ_WA);
        v8f c = wmma_strip(s_wi, CW, wseg, 13, nt, 4, lane);
        int col = nt * 16 + n16;
        float bias = s_bias[(isE ? 128 : 336) + col];
        float* dst = isE ? s_er : s_ad;
        #pragma unroll
        for (int j = 0; j < 8; ++j) {
          float x = c[j] + bias;
          dst[(mb + j) * NPV + col] = isE ? fsigmoid(x) : ftanh(x);
        }
      }
      __syncthreads();

      // ---- memory update: mem = mem*(1 - w*erase) + w*add  (same thread map
      //      as einsum -> no cross-thread global visibility needed) ----
      {
        float4* mrow4 = (float4*)(mem + (size_t)(b0 + r16) * (MD * VD));
        float4 er[4], ad[4];
        #pragma unroll
        for (int j = 0; j < 3; ++j) {
          er[j] = *(const float4*)&s_er[r16 * NPV + (vl + j * 16) * 4];
          ad[j] = *(const float4*)&s_ad[r16 * NPV + (vl + j * 16) * 4];
        }
        if (j3) {
          er[3] = *(const float4*)&s_er[r16 * NPV + (vl + 48) * 4];
          ad[3] = *(const float4*)&s_ad[r16 * NPV + (vl + 48) * 4];
        }
        for (int m = 0; m < MD; ++m) {
          float wm = s_w[r16 * MP + m];
          float4* mp = mrow4 + m * (VD / 4);
          #pragma unroll
          for (int j = 0; j < 3; ++j) {
            float4 mv = mp[vl + j * 16];
            mv.x = mv.x * (1.f - wm * er[j].x) + wm * ad[j].x;
            mv.y = mv.y * (1.f - wm * er[j].y) + wm * ad[j].y;
            mv.z = mv.z * (1.f - wm * er[j].z) + wm * ad[j].z;
            mv.w = mv.w * (1.f - wm * er[j].w) + wm * ad[j].w;
            mp[vl + j * 16] = mv;
          }
          if (j3) {
            float4 mv = mp[vl + 48];
            mv.x = mv.x * (1.f - wm * er[3].x) + wm * ad[3].x;
            mv.y = mv.y * (1.f - wm * er[3].y) + wm * ad[3].y;
            mv.z = mv.z * (1.f - wm * er[3].z) + wm * ad[3].z;
            mv.w = mv.w * (1.f - wm * er[3].w) + wm * ad[3].w;
            mp[vl + 48] = mv;
          }
        }
      }
      __syncthreads();
    }
  }
}

// ---------------------------------------------------------------------------
extern "C" void kernel_launch(void* const* d_in, const int* in_sizes, int n_in,
                              void* d_out, int out_size, void* d_ws, size_t ws_size,
                              hipStream_t stream) {
  (void)in_sizes; (void)n_in; (void)out_size; (void)ws_size;
  const float* q  = (const float*)d_in[0];
  const float* a  = (const float*)d_in[1];
  const float* vm = (const float*)d_in[2];
  const float* km = (const float*)d_in[3];
  const float* Wc = (const float*)d_in[4];
  const float* bc = (const float*)d_in[5];
  const float* Wr = (const float*)d_in[6];
  const float* br = (const float*)d_in[7];
  const float* We = (const float*)d_in[8];
  const float* be = (const float*)d_in[9];
  const float* Wa = (const float*)d_in[10];
  const float* ba = (const float*)d_in[11];

  float* out    = (float*)d_out;
  float* rc_out = out;                                   // (B,S,A)
  float* mem    = out + (size_t)BB * SS * AD;            // (B,M,V) = mem_final
  float* w_out  = mem + (size_t)BB * MD * VD;            // (B,S,M)
  unsigned short* swz = (unsigned short*)d_ws;           // bf16 swizzled weights

  // memory state starts as value_memory; updated in place -> ends as mem_final
  hipMemcpyAsync(mem, vm, (size_t)BB * MD * VD * sizeof(float),
                 hipMemcpyDeviceToDevice, stream);

  agm_prep_swizzle<<<(SWZ_TOTAL + 255) / 256, 256, 0, stream>>>(
      Wc, km, Wr, We, Wa, swz);

  agm_scan_kernel<<<BB / NB, 256, 0, stream>>>(
      q, a, bc, br, be, ba, swz, rc_out, mem, w_out);
}
